// GroupedQueryAttention_36790689858225
// MI455X (gfx1250) — matrix operand order repaired and checked by reading, not compiled, not run on hardware
//
#include <hip/hip_runtime.h>
#include <hip/hip_bf16.h>
#include <math.h>
#include <stdint.h>

#define DEV __device__ __forceinline__

typedef __bf16 v16bf __attribute__((ext_vector_type(16)));
typedef float  v8f   __attribute__((ext_vector_type(8)));
typedef unsigned v4u __attribute__((ext_vector_type(4)));
typedef int      v8i __attribute__((ext_vector_type(8)));
typedef int      v4i __attribute__((ext_vector_type(4)));

union Frag { unsigned u[8]; v16bf v; };

DEV unsigned short f2bf(float x) {
  union { float f; unsigned u; } c; c.f = x;
  unsigned r = (c.u + 0x7FFFu + ((c.u >> 16) & 1u)) >> 16;
  return (unsigned short)r;
}
DEV unsigned pack2(float a, float b) {
  return (unsigned)f2bf(a) | ((unsigned)f2bf(b) << 16);
}

DEV v8f wmma_bf16(v16bf a, v16bf b, v8f c) {
  return __builtin_amdgcn_wmma_f32_16x16x32_bf16(false, a, false, b, (short)0, c, false, false);
}

// ---------------- TDM support (Tensor Data Mover) ----------------
#if defined(__has_builtin)
#if __has_builtin(__builtin_amdgcn_tensor_load_to_lds) && \
    __has_builtin(__builtin_amdgcn_s_wait_tensorcnt)
#define HAVE_TDM 1
#endif
#endif
#ifndef HAVE_TDM
#define HAVE_TDM 0
#endif

#if HAVE_TDM
// 2-D bf16 tile load: tile_rows rows of tile_k elements, row stride = k_stride
// elements; LDS padded 1 dword per 16 dwords -> [rows][17] dword layout.
DEV void tdm_load_2d(const void* gptr, unsigned lds_off, unsigned k_stride,
                     unsigned tile_rows, unsigned tile_k) {
  uint64_t ga = (uint64_t)(uintptr_t)gptr;
  v4u g0;
  g0.x = 1u;                                   // count=1, user descriptor
  g0.y = lds_off;                              // LDS byte offset
  g0.z = (unsigned)ga;                         // global_addr[31:0]
  g0.w = (unsigned)((ga >> 32) & 0x01ffffffu) | (2u << 30);  // addr[56:32], type=2
  unsigned dim0 = k_stride;                    // tensor_dim0 (>= tile_k: no OOB)
  unsigned dim1 = 0x40000000u;                 // tensor_dim1 large (no OOB)
  v8i g1;
  g1[0] = (int)((1u << 16) | (1u << 20) | (3u << 22));  // 2B data, pad_en, 16DW/+1DW
  g1[1] = (int)((dim0 & 0xffffu) << 16);                // dim0[15:0]
  g1[2] = (int)((dim0 >> 16) | ((dim1 & 0xffffu) << 16));
  g1[3] = (int)((dim1 >> 16) | (tile_k << 16));         // tile_dim0
  g1[4] = (int)tile_rows;                               // tile_dim1 (tile_dim2=0)
  g1[5] = (int)k_stride;                                // tensor_dim0_stride[31:0]
  g1[6] = 0;
  g1[7] = 0;
  v4i z4 = {0, 0, 0, 0};
#if __clang_major__ >= 23
  v8i z8 = {0, 0, 0, 0, 0, 0, 0, 0};
  __builtin_amdgcn_tensor_load_to_lds(g0, g1, z4, z4, z8, 0);
#else
  __builtin_amdgcn_tensor_load_to_lds(g0, g1, z4, z4, 0);
#endif
}
#endif

// ---------------------------------------------------------------------------
// One-time conversions: f32 -> packed bf16 pairs; weights also transposed to
// [N][K] so every WMMA B-fragment pair (along K) is one contiguous u32.
// ---------------------------------------------------------------------------
__global__ void cvt_bf16_k(const float* __restrict__ in, unsigned* __restrict__ outp,
                           size_t npairs) {
  size_t i = (size_t)blockIdx.x * blockDim.x + threadIdx.x;
  if (i < npairs) {
    float2 t = *(const float2*)(in + 2 * i);
    outp[i] = pack2(t.x, t.y);
  }
}

__global__ void cvtT_bf16_k(const float* __restrict__ W, unsigned* __restrict__ WT,
                            int KP /*K/2*/, int N, size_t npairs) {
  size_t i = (size_t)blockIdx.x * blockDim.x + threadIdx.x;
  if (i >= npairs) return;
  int kp = (int)(i % KP);
  int n  = (int)(i / KP);
  WT[i] = pack2(W[(size_t)(2 * kp) * N + n], W[(size_t)(2 * kp + 1) * N + n]);
}

// ---------------------------------------------------------------------------
// GEMM: C[M,N](f32) = A[M,K](bf16) @ B^T[N,K](bf16), bf16 WMMA, f32 accum.
// 64x64 tile per WG (128 thr = 4 waves, each 32x32), BK=32.
// TDM path: double-buffered tensor_load_to_lds (padded [64][17] dword tiles),
// wave0 issues DMA for stage j+1 before computing stage j (TENSORcnt in-order).
// ---------------------------------------------------------------------------
__global__ __launch_bounds__(128) void gemm_bf16_k(const unsigned* __restrict__ A32,
                                                   const unsigned* __restrict__ BT32,
                                                   float* __restrict__ C,
                                                   int M, int N, int K) {
  const int tid  = threadIdx.x;
  const int lane = tid & 31, lm = lane & 15, lhi = lane >> 4;
  const int wv   = tid >> 5;
  const int wm   = (wv >> 1) * 32, wn = (wv & 1) * 32;
  const int row0 = blockIdx.y * 64, col0 = blockIdx.x * 64;
  const int KP   = K >> 1;

  v8f acc[2][2] = {};

#if HAVE_TDM
  __shared__ unsigned lA[2][64 * 17];
  __shared__ unsigned lB[2][64 * 17];
  const unsigned short* Ab = (const unsigned short*)A32;
  const unsigned short* Bb = (const unsigned short*)BT32;
  if (wv == 0) {
    tdm_load_2d(Ab + (size_t)row0 * K, (unsigned)(uintptr_t)&lA[0][0], K, 64, 32);
    tdm_load_2d(Bb + (size_t)col0 * K, (unsigned)(uintptr_t)&lB[0][0], K, 64, 32);
  }
  for (int k0 = 0; k0 < K; k0 += 32) {
    const int buf = (k0 >> 5) & 1;
    if (wv == 0) {
      if (k0 + 32 < K) {
        tdm_load_2d(Ab + (size_t)row0 * K + k0 + 32,
                    (unsigned)(uintptr_t)&lA[buf ^ 1][0], K, 64, 32);
        tdm_load_2d(Bb + (size_t)col0 * K + k0 + 32,
                    (unsigned)(uintptr_t)&lB[buf ^ 1][0], K, 64, 32);
        __builtin_amdgcn_s_wait_tensorcnt(2);  // stage j done, j+1 in flight
      } else {
        __builtin_amdgcn_s_wait_tensorcnt(0);
      }
    }
    __syncthreads();
    Frag a[2], b[2];
#pragma unroll
    for (int mt = 0; mt < 2; ++mt)
#pragma unroll
      for (int v = 0; v < 8; ++v) {
        int kp = lhi * 8 + v;  // A 16x32 bf16 layout
        a[mt].u[v] = lA[buf][(wm + mt * 16 + lm) * 17 + kp];
      }
#pragma unroll
    for (int nt = 0; nt < 2; ++nt)
#pragma unroll
      for (int v = 0; v < 8; ++v)  // B 32x16 bf16 layout
        b[nt].u[v] = lB[buf][(wn + nt * 16 + lm) * 17 + lhi * 8 + v];
#pragma unroll
    for (int mt = 0; mt < 2; ++mt)
#pragma unroll
      for (int nt = 0; nt < 2; ++nt)
        acc[mt][nt] = wmma_bf16(a[mt].v, b[nt].v, acc[mt][nt]);
    __syncthreads();
  }
#else
  // Fallback: fragments straight from global (pairs along K contiguous; L2-resident)
  for (int k0 = 0; k0 < K; k0 += 32) {
    Frag a[2], b[2];
#pragma unroll
    for (int mt = 0; mt < 2; ++mt)
#pragma unroll
      for (int v = 0; v < 8; ++v) {
        int k = k0 + lhi * 16 + 2 * v;
        a[mt].u[v] = A32[(size_t)(row0 + wm + mt * 16 + lm) * KP + (k >> 1)];
      }
#pragma unroll
    for (int nt = 0; nt < 2; ++nt)
#pragma unroll
      for (int v = 0; v < 8; ++v) {
        int k = k0 + lhi * 16 + 2 * v;
        b[nt].u[v] = BT32[(size_t)(col0 + wn + nt * 16 + lm) * KP + (k >> 1)];
      }
#pragma unroll
    for (int mt = 0; mt < 2; ++mt)
#pragma unroll
      for (int nt = 0; nt < 2; ++nt)
        acc[mt][nt] = wmma_bf16(a[mt].v, b[nt].v, acc[mt][nt]);
  }
#endif
  // C/D layout: lane holds col n=lm, VGPR r holds row r + 8*lhi
#pragma unroll
  for (int mt = 0; mt < 2; ++mt)
#pragma unroll
    for (int nt = 0; nt < 2; ++nt)
#pragma unroll
      for (int r = 0; r < 8; ++r) {
        int row = row0 + wm + mt * 16 + lhi * 8 + r;
        int col = col0 + wn + nt * 16 + lm;
        C[(size_t)row * N + col] = acc[mt][nt][r];
      }
}

// ---------------------------------------------------------------------------
// RMSNorm + RoPE for one (b, s, head) per 128-thread block; bf16 output.
// ---------------------------------------------------------------------------
__global__ __launch_bounds__(128) void qk_post_k(const float* __restrict__ in,
                                                 int row_stride, int head_stride,
                                                 const float* __restrict__ w,
                                                 const float* __restrict__ cosT,
                                                 const float* __restrict__ sinT,
                                                 unsigned short* __restrict__ outp,
                                                 int NH, int S_) {
  __shared__ float sh[128];
  __shared__ float red[4];
  int bid = blockIdx.x;
  int h = bid % NH; int bs = bid / NH;
  int s = bs % S_;  int b  = bs / S_;
  int d = threadIdx.x;
  float x = in[(size_t)(b * S_ + s) * row_stride + h * head_stride + d];
  float ss = x * x;
#pragma unroll
  for (int off = 16; off; off >>= 1) ss += __shfl_xor(ss, off, 32);
  if ((threadIdx.x & 31) == 0) red[threadIdx.x >> 5] = ss;
  __syncthreads();
  float tot = red[0] + red[1] + red[2] + red[3];
  float n = x * rsqrtf(tot * (1.0f / 128.0f) + 1e-6f) * (1.0f + w[d]);
  sh[d] = n;
  __syncthreads();
  float rot = (d < 64) ? -sh[d + 64] : sh[d - 64];
  float v = n * cosT[s * 128 + d] + rot * sinT[s * 128 + d];
  outp[((size_t)(b * NH + h) * S_ + s) * 128 + d] = f2bf(v);
}

// V: [b,s,g*128+d] f32 -> Vt [b,g,d,s] bf16 (P@V B-frag pairs contiguous)
__global__ void vtrans_k(const float* __restrict__ Vp,
                         unsigned short* __restrict__ Vt, int S_) {
  size_t idx = (size_t)blockIdx.x * blockDim.x + threadIdx.x;
  int s = (int)(idx % S_);
  size_t t = idx / S_;
  int d = (int)(t % 128);
  int bg = (int)(t / 128);
  int b = bg >> 2, g = bg & 3;
  Vt[idx] = f2bf(Vp[(size_t)(b * S_ + s) * 512 + g * 128 + d]);
}

// ---------------------------------------------------------------------------
// Flash attention: one wave per 16-query tile; 32 keys/iter.
// 8 WMMAs for scores + 8 WMMAs for ctx per iteration; online softmax with
// half-wave shfl reductions; P re-laid out C->A via per-wave LDS bounce.
// Epilogue fuses 1/l, sigmoid(gate), bf16 store.
// ---------------------------------------------------------------------------
__global__ __launch_bounds__(256, 1) void attn_k(const unsigned* __restrict__ Qb,
                                                 const unsigned* __restrict__ Kb,
                                                 const unsigned* __restrict__ Vt,
                                                 const float* __restrict__ QG,
                                                 unsigned short* __restrict__ CtxG) {
  constexpr int S_ = 2048;
  constexpr float SC = 0.08838834764831845f * 1.44269504088896340f;  // HD^-0.5*log2e
  __shared__ unsigned short Pls[8][16 * 32];

  const int tid = threadIdx.x, lane = tid & 31, lm = lane & 15, lhi = lane >> 4;
  const int wv = tid >> 5;
  unsigned short* pP = &Pls[wv][0];
  unsigned* pP32 = (unsigned*)pP;

  int w  = blockIdx.x * 8 + wv;
  int qt = w & 127;
  int h  = (w >> 7) & 15;
  int b  = w >> 11;
  int q0 = qt * 16;
  int bh = b * 16 + h;
  int bg = b * 4 + (h >> 2);

  Frag qf[4];
#pragma unroll
  for (int kc = 0; kc < 4; ++kc)
#pragma unroll
    for (int v = 0; v < 8; ++v) {
      int d = kc * 32 + lhi * 16 + 2 * v;
      qf[kc].u[v] = Qb[(((size_t)bh * S_ + q0 + lm) * 128 + d) >> 1];
    }

  float mrow[8], lrow[8];
  v8f o[8] = {};
#pragma unroll
  for (int r = 0; r < 8; ++r) { mrow[r] = -1e30f; lrow[r] = 0.0f; }

  for (int k0 = 0; k0 < q0 + 16; k0 += 32) {
    v8f sc[2] = {};
#pragma unroll
    for (int kt = 0; kt < 2; ++kt)
#pragma unroll
      for (int kc = 0; kc < 4; ++kc) {
        Frag kf;
#pragma unroll
        for (int v = 0; v < 8; ++v) {
          int d = kc * 32 + lhi * 16 + 2 * v;
          kf.u[v] = Kb[(((size_t)bg * S_ + k0 + kt * 16 + lm) * 128 + d) >> 1];
        }
        sc[kt] = wmma_bf16(qf[kc].v, kf.v, sc[kt]);
      }

    const bool need_mask = (k0 + 31 > q0);
    float af[8];
#pragma unroll
    for (int r = 0; r < 8; ++r) {
      int m = lhi * 8 + r;
      float s0 = sc[0][r] * SC;
      float s1 = sc[1][r] * SC;
      if (need_mask) {
        if (k0 + lm      > q0 + m) s0 = -1e30f;
        if (k0 + 16 + lm > q0 + m) s1 = -1e30f;
      }
      float t = fmaxf(s0, s1);
#pragma unroll
      for (int off = 8; off; off >>= 1) t = fmaxf(t, __shfl_xor(t, off, 32));
      float mn = fmaxf(mrow[r], t);
      float al = exp2f(mrow[r] - mn);
      float p0 = exp2f(s0 - mn), p1 = exp2f(s1 - mn);
      float rs = p0 + p1;
#pragma unroll
      for (int off = 8; off; off >>= 1) rs += __shfl_xor(rs, off, 32);
      lrow[r] = lrow[r] * al + rs;
      mrow[r] = mn;
      af[r] = al;
      pP[m * 32 + lm]      = f2bf(p0);
      pP[m * 32 + 16 + lm] = f2bf(p1);
    }
    Frag pf;
#pragma unroll
    for (int v = 0; v < 8; ++v) {
      int kk = lhi * 16 + 2 * v;
      pf.u[v] = pP32[(lm * 32 + kk) >> 1];
    }
#pragma unroll
    for (int nt = 0; nt < 8; ++nt) {
      Frag vf;
#pragma unroll
      for (int v = 0; v < 8; ++v) {
        int key = k0 + lhi * 16 + 2 * v;
        vf.u[v] = Vt[(((size_t)bg * 128 + nt * 16 + lm) * S_ + key) >> 1];
      }
#pragma unroll
      for (int r = 0; r < 8; ++r) o[nt][r] *= af[r];
      o[nt] = wmma_bf16(pf.v, vf.v, o[nt]);
    }
  }

#pragma unroll
  for (int r = 0; r < 8; ++r) lrow[r] = 1.0f / lrow[r];
#pragma unroll
  for (int nt = 0; nt < 8; ++nt)
#pragma unroll
    for (int r = 0; r < 8; ++r) {
      int srow = q0 + lhi * 8 + r;
      int d = nt * 16 + lm;
      float val = o[nt][r] * lrow[r];
      float gt = QG[(size_t)(b * S_ + srow) * 4096 + h * 256 + 128 + d];
      val *= 1.0f / (1.0f + exp2f(-gt * 1.44269504088896340f));
      CtxG[(size_t)(b * S_ + srow) * 2048 + h * 128 + d] = f2bf(val);
    }
}

// ---------------------------------------------------------------------------
extern "C" void kernel_launch(void* const* d_in, const int* in_sizes, int n_in,
                              void* d_out, int out_size, void* d_ws, size_t ws_size,
                              hipStream_t stream) {
  constexpr int Bb = 2, Ss = 2048, Din = 2048, Hh = 16, Gg = 4, HD = 128;
  constexpr int Dout = Hh * HD;   // 2048
  constexpr int M = Bb * Ss;      // 4096

  const float* x    = (const float*)d_in[0];
  const float* Wq   = (const float*)d_in[1];
  const float* Wk   = (const float*)d_in[2];
  const float* Wv   = (const float*)d_in[3];
  const float* Wo   = (const float*)d_in[4];
  const float* qw   = (const float*)d_in[5];
  const float* kw   = (const float*)d_in[6];
  const float* cosT = (const float*)d_in[7];
  const float* sinT = (const float*)d_in[8];
  // d_in[9] = mask: causal, computed analytically in-kernel
  float* out = (float*)d_out;
  (void)in_sizes; (void)n_in; (void)out_size; (void)ws_size;

  char* ws = (char*)d_ws;
  size_t off = 0;
  auto alloc = [&](size_t bytes) -> void* {
    void* p = (void*)(ws + off);
    off += (bytes + 255) & ~(size_t)255;
    return p;
  };
  unsigned* xb32  = (unsigned*)alloc((size_t)M * Din / 2 * 4);          // x bf16
  unsigned* WqT32 = (unsigned*)alloc((size_t)(2 * Dout) * Din / 2 * 4); // Wq^T bf16
  unsigned* WkT32 = (unsigned*)alloc((size_t)(Gg * HD) * Din / 2 * 4);
  unsigned* WvT32 = (unsigned*)alloc((size_t)(Gg * HD) * Din / 2 * 4);
  unsigned* WoT32 = (unsigned*)alloc((size_t)Din * Dout / 2 * 4);       // Wo^T bf16
  float* QG           = (float*)alloc((size_t)M * 2 * Dout * 4);
  float* Kp           = (float*)alloc((size_t)M * Gg * HD * 4);
  float* Vp           = (float*)alloc((size_t)M * Gg * HD * 4);
  unsigned short* Qb  = (unsigned short*)alloc((size_t)Bb * Hh * Ss * HD * 2);
  unsigned short* Kb  = (unsigned short*)alloc((size_t)Bb * Gg * Ss * HD * 2);
  unsigned short* Vt  = (unsigned short*)alloc((size_t)Bb * Gg * HD * Ss * 2);
  unsigned short* Ctx = (unsigned short*)alloc((size_t)M * Dout * 2);

  // 0) one-time bf16 conversions (weights transposed to [N][K])
  {
    size_t np;
    np = (size_t)M * Din / 2;
    cvt_bf16_k<<<(unsigned)((np + 255) / 256), 256, 0, stream>>>(x, xb32, np);
    np = (size_t)(2 * Dout) * Din / 2;
    cvtT_bf16_k<<<(unsigned)((np + 255) / 256), 256, 0, stream>>>(Wq, WqT32, Din / 2, 2 * Dout, np);
    np = (size_t)(Gg * HD) * Din / 2;
    cvtT_bf16_k<<<(unsigned)((np + 255) / 256), 256, 0, stream>>>(Wk, WkT32, Din / 2, Gg * HD, np);
    cvtT_bf16_k<<<(unsigned)((np + 255) / 256), 256, 0, stream>>>(Wv, WvT32, Din / 2, Gg * HD, np);
    np = (size_t)Din * Dout / 2;
    cvtT_bf16_k<<<(unsigned)((np + 255) / 256), 256, 0, stream>>>(Wo, WoT32, Dout / 2, Din, np);
  }

  // 1) projections (bf16 WMMA, f32 accumulate; TDM-staged tiles)
  gemm_bf16_k<<<dim3(2 * Dout / 64, M / 64), 128, 0, stream>>>(xb32, WqT32, QG, M, 2 * Dout, Din);
  gemm_bf16_k<<<dim3(Gg * HD / 64, M / 64), 128, 0, stream>>>(xb32, WkT32, Kp, M, Gg * HD, Din);
  gemm_bf16_k<<<dim3(Gg * HD / 64, M / 64), 128, 0, stream>>>(xb32, WvT32, Vp, M, Gg * HD, Din);

  // 2) RMSNorm + RoPE -> bf16; V -> transposed bf16
  qk_post_k<<<Bb * Ss * Hh, 128, 0, stream>>>(QG, 2 * Dout, 2 * HD, qw, cosT, sinT, Qb, Hh, Ss);
  qk_post_k<<<Bb * Ss * Gg, 128, 0, stream>>>(Kp, Gg * HD, HD, kw, cosT, sinT, Kb, Gg, Ss);
  vtrans_k<<<(Bb * Gg * HD * Ss) / 256, 256, 0, stream>>>(Vp, Vt, Ss);

  // 3) flash attention + sigmoid gate (gate read directly from QG)
  attn_k<<<(Bb * Hh * (Ss / 16)) / 8, 256, 0, stream>>>(
      (const unsigned*)Qb, (const unsigned*)Kb, (const unsigned*)Vt, QG, Ctx);

  // 4) output projection
  gemm_bf16_k<<<dim3(Din / 64, M / 64), 128, 0, stream>>>(
      (const unsigned*)Ctx, WoT32, out, M, Din, Dout);
}
